// PairwiseFeatureExtractor_69286412419097
// MI455X (gfx1250) — compile-verified, matrix operand order checked by reading, not run on hardware
//
#include <hip/hip_runtime.h>
#include <hip/hip_bf16.h>

typedef __bf16 bf16;
typedef __bf16 v16bf __attribute__((ext_vector_type(16)));
typedef float  v8f   __attribute__((ext_vector_type(8)));

#define B_     16
#define O_     80
#define R_     512
#define IN_CH  4096
#define NCLS   151
#define EMB    768
#define HID    512
#define POOL   2048
#define GEO_   128
#define OBJCAT 4992   // IN_CH + EMB + GEO_
#define AUGW   5376   // EMB + IN_CH + HID

// gfx1250 async global->LDS path (guarded; falls back to register staging)
#if __has_builtin(__builtin_amdgcn_global_load_async_to_lds_b128) && \
    __has_builtin(__builtin_amdgcn_s_wait_asynccnt)
#define USE_ASYNC_LDS 1
typedef int v4i_vec __attribute__((vector_size(16)));   // b128 payload type
#else
#define USE_ASYNC_LDS 0
#endif

// ---------------------------------------------------------------------------
// f32 -> bf16 conversion (float4 in, packed 4x bf16 out) — for A-side tensors
// ---------------------------------------------------------------------------
__global__ void f32_to_bf16_kernel(const float* __restrict__ src,
                                   bf16* __restrict__ dst, long n4) {
  long i      = (long)blockIdx.x * blockDim.x + threadIdx.x;
  long stride = (long)gridDim.x * blockDim.x;
  for (; i < n4; i += stride) {
    float4 v = ((const float4*)src)[i];
    unsigned h0 = __builtin_bit_cast(unsigned short, (bf16)v.x);
    unsigned h1 = __builtin_bit_cast(unsigned short, (bf16)v.y);
    unsigned h2 = __builtin_bit_cast(unsigned short, (bf16)v.z);
    unsigned h3 = __builtin_bit_cast(unsigned short, (bf16)v.w);
    uint2 p;
    p.x = h0 | (h1 << 16);
    p.y = h2 | (h3 << 16);
    ((uint2*)dst)[i] = p;
  }
}

// ---------------------------------------------------------------------------
// Transposing f32 -> bf16 weight conversion: W[K,N] f32 -> WT[N,K] bf16.
// ---------------------------------------------------------------------------
__global__ void transpose_f32_to_bf16_kernel(const float* __restrict__ W,
                                             bf16* __restrict__ WT,
                                             int K, int N) {
  __shared__ float s[32][33];
  const int tx = threadIdx.x;         // 0..31
  const int ty = threadIdx.y;         // 0..7
  const int n0 = blockIdx.x * 32;
  const int k0 = blockIdx.y * 32;
  for (int j = 0; j < 4; j++) {
    const int kk = ty + j * 8;
    s[kk][tx] = W[(size_t)(k0 + kk) * N + n0 + tx];
  }
  __syncthreads();
  for (int j = 0; j < 4; j++) {
    const int nn = ty + j * 8;
    WT[(size_t)(n0 + nn) * K + k0 + tx] = (bf16)s[tx][nn];
  }
}

// ---------------------------------------------------------------------------
// Per-object prep: softmax @ emb_dist, pos MLP (9->32->BN->128, relu),
// roi copy, label-embedding gather.
// ---------------------------------------------------------------------------
__global__ void prep_obj_kernel(
    const float* __restrict__ roi,    const float* __restrict__ logits,
    const float* __restrict__ pos9,   const int*   __restrict__ labels,
    const float* __restrict__ emb_dist, const float* __restrict__ emb_label,
    const float* __restrict__ fc1w,   const float* __restrict__ fc1b,
    const float* __restrict__ bng,    const float* __restrict__ bnb,
    const float* __restrict__ bnm,    const float* __restrict__ bnv,
    const float* __restrict__ fc2w,   const float* __restrict__ fc2b,
    bf16* __restrict__ objcat, bf16* __restrict__ augment)
{
  __shared__ float s_prob[NCLS];
  __shared__ float s_red[2];
  __shared__ float s_h[32];
  const int i = blockIdx.x;
  const int t = threadIdx.x;

  if (t < NCLS) s_prob[t] = logits[(size_t)i * NCLS + t];
  __syncthreads();
  if (t == 0) {
    float m = -1e30f;
    for (int c = 0; c < NCLS; c++) m = fmaxf(m, s_prob[c]);
    s_red[0] = m;
  }
  __syncthreads();
  const float mx = s_red[0];
  if (t < NCLS) s_prob[t] = __expf(s_prob[t] - mx);
  __syncthreads();
  if (t == 0) {
    float s = 0.f;
    for (int c = 0; c < NCLS; c++) s += s_prob[c];
    s_red[1] = 1.0f / s;
  }
  __syncthreads();
  const float inv = s_red[1];

  for (int o = t; o < EMB; o += blockDim.x) {
    float acc = 0.f;
    for (int c = 0; c < NCLS; c++) acc += s_prob[c] * emb_dist[(size_t)c * EMB + o];
    objcat[(size_t)i * OBJCAT + IN_CH + o] = (bf16)(acc * inv);
  }
  for (int j = t; j < IN_CH; j += blockDim.x) {
    bf16 h = (bf16)roi[(size_t)i * IN_CH + j];
    objcat[(size_t)i * OBJCAT + j]      = h;
    augment[(size_t)i * AUGW + EMB + j] = h;
  }
  const int lbl = labels[i];
  for (int j = t; j < EMB; j += blockDim.x)
    augment[(size_t)i * AUGW + j] = (bf16)emb_label[(size_t)lbl * EMB + j];

  if (t < 32) {
    float h = fc1b[t];
    for (int k = 0; k < 9; k++) h += pos9[(size_t)i * 9 + k] * fc1w[k * 32 + t];
    h = (h - bnm[t]) * rsqrtf(bnv[t] + 1e-5f) * bng[t] + bnb[t];
    s_h[t] = h;
  }
  __syncthreads();
  if (t < GEO_) {
    float e = fc2b[t];
    for (int k = 0; k < 32; k++) e += s_h[k] * fc2w[k * GEO_ + t];
    objcat[(size_t)i * OBJCAT + IN_CH + EMB + t] = (bf16)fmaxf(e, 0.f);
  }
}

// ---------------------------------------------------------------------------
// Box-pair geometry (32 features) + spt1 (32 -> 512, relu), bf16 out
// ---------------------------------------------------------------------------
__device__ inline void box_info8(float x1, float y1, float x2, float y2, float* o) {
  o[0] = x1; o[1] = y1; o[2] = x2; o[3] = y2;
  o[4] = (x1 + x2) * 0.5f; o[5] = (y1 + y2) * 0.5f;
  o[6] = x2 - x1; o[7] = y2 - y1;
}

__global__ void geo_spt1_kernel(const float* __restrict__ boxes,
                                const int*   __restrict__ pairs,
                                const float* __restrict__ w1,
                                const float* __restrict__ b1v,
                                bf16* __restrict__ spt_hid)
{
  __shared__ float g[32];
  const int r = blockIdx.x;       // 0 .. B*R-1
  const int b = r / R_;
  const int t = threadIdx.x;
  if (t == 0) {
    const int i0 = pairs[(size_t)r * 2 + 0];
    const int i1 = pairs[(size_t)r * 2 + 1];
    const float* bx1 = boxes + ((size_t)b * O_ + i0) * 4;
    const float* bx2 = boxes + ((size_t)b * O_ + i1) * 4;
    box_info8(bx1[0], bx1[1], bx1[2], bx1[3], g + 0);
    box_info8(bx2[0], bx2[1], bx2[2], bx2[3], g + 8);
    float ux1 = fminf(bx1[0], bx2[0]), uy1 = fminf(bx1[1], bx2[1]);
    float ux2 = fmaxf(bx1[2], bx2[2]), uy2 = fmaxf(bx1[3], bx2[3]);
    box_info8(ux1, uy1, ux2, uy2, g + 16);
    float ix1 = fmaxf(bx1[0], bx2[0]), iy1 = fmaxf(bx1[1], bx2[1]);
    float ix2 = fminf(bx1[2], bx2[2]), iy2 = fminf(bx1[3], bx2[3]);
    float ii[8];
    box_info8(ix1, iy1, ix2, iy2, ii);
    float valid = (ix1 <= ix2 && iy1 <= iy2) ? 1.f : 0.f;
    for (int k = 0; k < 8; k++) g[24 + k] = ii[k] * valid;
  }
  __syncthreads();
  for (int o = t; o < HID; o += blockDim.x) {
    float acc = b1v[o];
    for (int k = 0; k < 32; k++) acc += g[k] * w1[k * HID + o];
    spt_hid[(size_t)r * HID + o] = (bf16)fmaxf(acc, 0.f);
  }
}

// ---------------------------------------------------------------------------
// Pair gather (hd|tl from fused) * spt -> bf16
// ---------------------------------------------------------------------------
__global__ void pair_mul_kernel(const float* __restrict__ fused,
                                const float* __restrict__ spt,
                                const int*   __restrict__ pairs,
                                bf16* __restrict__ out)
{
  const size_t n = (size_t)B_ * R_ * 2 * HID;
  for (size_t idx = (size_t)blockIdx.x * blockDim.x + threadIdx.x; idx < n;
       idx += (size_t)gridDim.x * blockDim.x) {
    const int    c    = (int)(idx % (2 * HID));
    const size_t r    = idx / (2 * HID);
    const int    b    = (int)(r / R_);
    const int    slot = c / HID;                 // 0: head, 1: tail
    const int    i    = pairs[r * 2 + slot];
    const float  f    = fused[((size_t)b * O_ + i) * (2 * HID) + c];
    out[idx] = (bf16)(f * spt[idx]);
  }
}

// ---------------------------------------------------------------------------
// Tiled bf16 WMMA GEMM: C = act(A[MxK] @ WT[NxK]^T + bias (+ Cin))
// 128x128 block tile, 256 threads = 8 waves (2M x 4N), wave tile 64x32
// (4x2 WMMA 16x16x32 frags).
// Pipeline: async global->LDS double buffer (ASYNCcnt) when the gfx1250
// builtins exist; else register-staged single LDS buffer.
// Requires M%128==0, N%128==0, K%32==0 (true for all calls here).
// ---------------------------------------------------------------------------
enum { GF_RELU = 1, GF_ADDC = 2, GF_OUTBF = 4 };

union FragBF { unsigned u[8]; v16bf v; };

#define LDS_STRIDE 40   // 32 + 8 pad (bf16 elems)
#define TILE_ELEMS (128 * LDS_STRIDE)

template <int FLAGS>
__global__ __launch_bounds__(256) void gemm_bf16_kernel(
    const bf16* __restrict__ A,  int lda,   // [M, K] K-major
    const bf16* __restrict__ BT, int ldbt,  // [N, K] K-major (W transposed)
    const float* __restrict__ bias,
    const float* __restrict__ Cin, int ldcin,
    float* __restrict__ Co, bf16* __restrict__ Cb, int ldc,
    int M, int N, int K)
{
  const int tid  = threadIdx.x;
  const int lane = tid & 31;
  const int wave = tid >> 5;
  const int wm   = (wave >> 2) * 64;       // wave tile origin (M): 0 / 64
  const int wn   = (wave & 3) * 32;        // wave tile origin (N): 0/32/64/96
  const int h    = lane >> 4;              // lane half (0/1)
  const int mn   = lane & 15;              // m (A) / n (B,C) within 16
  const int m0   = blockIdx.y * 128;
  const int n0   = blockIdx.x * 128;

  v8f acc[4][2];
  for (int a = 0; a < 4; a++)
    for (int bb = 0; bb < 2; bb++)
      for (int e = 0; e < 8; e++) acc[a][bb][e] = 0.f;

  // Cooperative tile mapping: 128 rows x 32 k, each thread owns 16 bf16.
  const int trow = tid >> 1;               // 0..127
  const int tcol = (tid & 1) * 16;         // 0 / 16

  const bf16* pA = A  + (size_t)(m0 + trow) * lda  + tcol;
  const bf16* pB = BT + (size_t)(n0 + trow) * ldbt + tcol;

#if USE_ASYNC_LDS
  // -------- async global->LDS, double-buffered over K --------
  __shared__ __align__(16) bf16 As[2][TILE_ELEMS];
  __shared__ __align__(16) bf16 Bs[2][TILE_ELEMS];

  bf16* dA0 = &As[0][trow * LDS_STRIDE + tcol];
  bf16* dB0 = &Bs[0][trow * LDS_STRIDE + tcol];
  bf16* dA1 = &As[1][trow * LDS_STRIDE + tcol];
  bf16* dB1 = &Bs[1][trow * LDS_STRIDE + tcol];

  // 4 async b128 ops per wave per tile (A lo/hi, B lo/hi)
  #define ISSUE_TILE(dA, dB, k)                                               \
    do {                                                                      \
      __builtin_amdgcn_global_load_async_to_lds_b128(                         \
          (v4i_vec*)(void*)(pA + (k)),     (v4i_vec*)(void*)(dA),     0, 0);  \
      __builtin_amdgcn_global_load_async_to_lds_b128(                         \
          (v4i_vec*)(void*)(pA + (k) + 8), (v4i_vec*)(void*)((dA) + 8), 0, 0);\
      __builtin_amdgcn_global_load_async_to_lds_b128(                         \
          (v4i_vec*)(void*)(pB + (k)),     (v4i_vec*)(void*)(dB),     0, 0);  \
      __builtin_amdgcn_global_load_async_to_lds_b128(                         \
          (v4i_vec*)(void*)(pB + (k) + 8), (v4i_vec*)(void*)((dB) + 8), 0, 0);\
    } while (0)

  ISSUE_TILE(dA0, dB0, 0);
  if (K > 32) ISSUE_TILE(dA1, dB1, 32);

  for (int k0 = 0; k0 < K; k0 += 32) {
    const int cur = (k0 >> 5) & 1;
    // in-order completion: <=4 outstanding means current tile has landed
    if (k0 + 32 < K) __builtin_amdgcn_s_wait_asynccnt(4);
    else             __builtin_amdgcn_s_wait_asynccnt(0);
    __syncthreads();

    const bf16* Ab = As[cur];
    const bf16* Bb = Bs[cur];
    FragBF af[4], bf[2];
    for (int a = 0; a < 4; a++) {
      const int mb = wm + a * 16 + mn;
      for (int v = 0; v < 8; v++) {
        const int kk = ((v & 3) << 1) + ((v >> 2) << 4) + (h << 3);
        af[a].u[v] = *(const unsigned*)&Ab[mb * LDS_STRIDE + kk];
      }
    }
    for (int bb = 0; bb < 2; bb++) {
      const int nb = wn + bb * 16 + mn;
      for (int v = 0; v < 8; v++) {
        const int kk = ((v & 3) << 1) + ((v >> 2) << 4) + (h << 3);
        bf[bb].u[v] = *(const unsigned*)&Bb[nb * LDS_STRIDE + kk];
      }
    }
    for (int a = 0; a < 4; a++)
      for (int bb = 0; bb < 2; bb++)
        acc[a][bb] = __builtin_amdgcn_wmma_f32_16x16x32_bf16(
            false, af[a].v, false, bf[bb].v, (short)0, acc[a][bb],
            false, false);
    __syncthreads();   // everyone done reading buffer `cur`

    if (k0 + 64 < K) {
      if (cur == 0) ISSUE_TILE(dA0, dB0, k0 + 64);
      else          ISSUE_TILE(dA1, dB1, k0 + 64);
    }
  }
  #undef ISSUE_TILE
#else
  // -------- register-staged single LDS buffer --------
  __shared__ __align__(16) bf16 As[TILE_ELEMS];
  __shared__ __align__(16) bf16 Bs[TILE_ELEMS];

  uint4 ra0 = ((const uint4*)pA)[0];
  uint4 ra1 = ((const uint4*)pA)[1];
  uint4 rb0 = ((const uint4*)pB)[0];
  uint4 rb1 = ((const uint4*)pB)[1];

  for (int k0 = 0; k0 < K; k0 += 32) {
    uint4* dA = (uint4*)&As[trow * LDS_STRIDE + tcol];
    uint4* dB = (uint4*)&Bs[trow * LDS_STRIDE + tcol];
    dA[0] = ra0; dA[1] = ra1;
    dB[0] = rb0; dB[1] = rb1;
    __syncthreads();

    if (k0 + 32 < K) {
      ra0 = ((const uint4*)(pA + k0 + 32))[0];
      ra1 = ((const uint4*)(pA + k0 + 32))[1];
      rb0 = ((const uint4*)(pB + k0 + 32))[0];
      rb1 = ((const uint4*)(pB + k0 + 32))[1];
    }
    if (k0 + 64 < K) {
      __builtin_prefetch(pA + k0 + 64, 0, 1);
      __builtin_prefetch(pB + k0 + 64, 0, 1);
    }

    FragBF af[4], bf[2];
    for (int a = 0; a < 4; a++) {
      const int mb = wm + a * 16 + mn;
      for (int v = 0; v < 8; v++) {
        const int kk = ((v & 3) << 1) + ((v >> 2) << 4) + (h << 3);
        af[a].u[v] = *(const unsigned*)&As[mb * LDS_STRIDE + kk];
      }
    }
    for (int bb = 0; bb < 2; bb++) {
      const int nb = wn + bb * 16 + mn;
      for (int v = 0; v < 8; v++) {
        const int kk = ((v & 3) << 1) + ((v >> 2) << 4) + (h << 3);
        bf[bb].u[v] = *(const unsigned*)&Bs[nb * LDS_STRIDE + kk];
      }
    }
    for (int a = 0; a < 4; a++)
      for (int bb = 0; bb < 2; bb++)
        acc[a][bb] = __builtin_amdgcn_wmma_f32_16x16x32_bf16(
            false, af[a].v, false, bf[bb].v, (short)0, acc[a][bb],
            false, false);
    __syncthreads();
  }
#endif

  // Epilogue: C 16x16 f32 layout — VGPR r: row = r + 8*h, col = lane&15
  for (int a = 0; a < 4; a++)
    for (int bb = 0; bb < 2; bb++) {
      const int col = n0 + wn + bb * 16 + mn;
      for (int r = 0; r < 8; r++) {
        const int row = m0 + wm + a * 16 + h * 8 + r;
        float v = acc[a][bb][r] + bias[col];
        if (FLAGS & GF_ADDC) v += Cin[(size_t)row * ldcin + col];
        if (FLAGS & GF_RELU) v = fmaxf(v, 0.f);
        if (FLAGS & GF_OUTBF) Cb[(size_t)row * ldc + col] = (bf16)v;
        else                  Co[(size_t)row * ldc + col] = v;
      }
    }
}

// ---------------------------------------------------------------------------
// Host side
// ---------------------------------------------------------------------------
extern "C" void kernel_launch(void* const* d_in, const int* in_sizes, int n_in,
                              void* d_out, int out_size, void* d_ws, size_t ws_size,
                              hipStream_t stream)
{
  const float* roi      = (const float*)d_in[0];
  const float* unionf   = (const float*)d_in[1];
  const float* logits   = (const float*)d_in[2];
  const float* pos9     = (const float*)d_in[3];
  const float* boxes    = (const float*)d_in[4];
  const int*   labels   = (const int*)  d_in[5];
  const int*   pairs    = (const int*)  d_in[6];
  const float* emb_dist = (const float*)d_in[7];
  const float* emb_label= (const float*)d_in[8];
  const float* relup_w  = (const float*)d_in[9];
  const float* relup_b  = (const float*)d_in[10];
  const float* updim_w  = (const float*)d_in[11];
  const float* updim_b  = (const float*)d_in[12];
  const float* fc1w     = (const float*)d_in[13];
  const float* fc1b     = (const float*)d_in[14];
  const float* bng      = (const float*)d_in[15];
  const float* bnb      = (const float*)d_in[16];
  const float* bnm      = (const float*)d_in[17];
  const float* bnv      = (const float*)d_in[18];
  const float* fc2w     = (const float*)d_in[19];
  const float* fc2b     = (const float*)d_in[20];
  const float* spt1w    = (const float*)d_in[21];
  const float* spt1b    = (const float*)d_in[22];
  const float* spt2w    = (const float*)d_in[23];
  const float* spt2b    = (const float*)d_in[24];
  const float* finw     = (const float*)d_in[25];
  const float* finb     = (const float*)d_in[26];
  const float* objhidw  = (const float*)d_in[27];
  const float* objhidb  = (const float*)d_in[28];
  const float* augw     = (const float*)d_in[29];
  const float* augb     = (const float*)d_in[30];
  (void)in_sizes; (void)n_in; (void)out_size; (void)ws_size;

  char*  ws  = (char*)d_ws;
  size_t off = 0;
  auto alloc = [&](size_t bytes) -> char* {
    char* p = ws + off;
    off = (off + bytes + 255) & ~(size_t)255;
    return p;
  };

  bf16*  objcat    = (bf16*) alloc((size_t)1280 * OBJCAT * 2);
  bf16*  augment   = (bf16*) alloc((size_t)1280 * AUGW   * 2);
  float* fused     = (float*)alloc((size_t)1280 * 1024   * 4);
  bf16*  spt_hid   = (bf16*) alloc((size_t)8192 * HID    * 2);
  float* spt       = (float*)alloc((size_t)8192 * 1024   * 4);
  bf16*  pairmul   = (bf16*) alloc((size_t)8192 * 1024   * 2);
  bf16*  wt_objhid = (bf16*) alloc((size_t)OBJCAT * 512  * 2);  // [512,  OBJCAT]
  bf16*  wt_updim  = (bf16*) alloc((size_t)AUGW * 1024   * 2);  // [1024, AUGW]
  bf16*  wt_aug    = (bf16*) alloc((size_t)AUGW * 2048   * 2);  // [2048, AUGW]
  bf16*  wt_spt2   = (bf16*) alloc((size_t)512  * 1024   * 2);  // [1024, 512]
  bf16*  wt_fin    = (bf16*) alloc((size_t)1024 * 2048   * 2);  // [2048, 1024]
  bf16*  wt_relup  = (bf16*) alloc((size_t)4096 * 2048   * 2);  // [2048, 4096]
  bf16*  unionbf   = (bf16*) alloc((size_t)8192 * 4096   * 2);

  auto tconv = [&](const float* s, bf16* d, int K, int N) {
    transpose_f32_to_bf16_kernel<<<dim3(N / 32, K / 32), dim3(32, 8), 0, stream>>>(
        s, d, K, N);
  };
  tconv(objhidw, wt_objhid, OBJCAT, 512);
  tconv(updim_w, wt_updim,  AUGW,   1024);
  tconv(augw,    wt_aug,    AUGW,   2048);
  tconv(spt2w,   wt_spt2,   512,    1024);
  tconv(finw,    wt_fin,    1024,   2048);
  tconv(relup_w, wt_relup,  4096,   2048);

  {
    long n4 = (long)((size_t)8192 * 4096 / 4);
    f32_to_bf16_kernel<<<4096, 256, 0, stream>>>(unionf, unionbf, n4);
  }

  prep_obj_kernel<<<1280, 256, 0, stream>>>(
      roi, logits, pos9, labels, emb_dist, emb_label,
      fc1w, fc1b, bng, bnb, bnm, bnv, fc2w, fc2b, objcat, augment);

  float* out_aug = (float*)d_out;                      // [1280, 2048]
  float* out_rel = out_aug + (size_t)1280 * 2048;      // [8192, 2048]

  // aug_hid = objcat @ objhid_w + b  -> bf16 into augment cols [4864,5376)
  gemm_bf16_kernel<GF_OUTBF><<<dim3(512 / 128, 1280 / 128), 256, 0, stream>>>(
      objcat, OBJCAT, wt_objhid, OBJCAT, objhidb,
      nullptr, 0, nullptr, augment + (AUGW - HID), AUGW, 1280, 512, OBJCAT);

  // fused = augment @ updim_w + b   (f32, [1280, 1024])
  gemm_bf16_kernel<0><<<dim3(1024 / 128, 1280 / 128), 256, 0, stream>>>(
      augment, AUGW, wt_updim, AUGW, updim_b,
      nullptr, 0, fused, nullptr, 1024, 1280, 1024, AUGW);

  // augment_out = relu(augment @ aug_w + b)  -> d_out part 1
  gemm_bf16_kernel<GF_RELU><<<dim3(2048 / 128, 1280 / 128), 256, 0, stream>>>(
      augment, AUGW, wt_aug, AUGW, augb,
      nullptr, 0, out_aug, nullptr, 2048, 1280, 2048, AUGW);

  // geometry + spt1 (32 -> 512, relu), bf16
  geo_spt1_kernel<<<B_ * R_, 256, 0, stream>>>(boxes, pairs, spt1w, spt1b, spt_hid);

  // spt = relu(spt_hid @ spt2_w + b)  (f32, [8192, 1024])
  gemm_bf16_kernel<GF_RELU><<<dim3(1024 / 128, 8192 / 128), 256, 0, stream>>>(
      spt_hid, HID, wt_spt2, HID, spt2b,
      nullptr, 0, spt, nullptr, 1024, 8192, 1024, HID);

  // pair gather * spt -> bf16 [8192, 1024]
  pair_mul_kernel<<<4096, 256, 0, stream>>>(fused, spt, pairs, pairmul);

  // rel = relu(pairmul @ fin_w + b) -> d_out part 2
  gemm_bf16_kernel<GF_RELU><<<dim3(2048 / 128, 8192 / 128), 256, 0, stream>>>(
      pairmul, 1024, wt_fin, 1024, finb,
      nullptr, 0, out_rel, nullptr, 2048, 8192, 2048, 1024);

  // rel += union @ relup_w + relup_b  (the 137 GFLOP GEMM)
  gemm_bf16_kernel<GF_ADDC><<<dim3(2048 / 128, 8192 / 128), 256, 0, stream>>>(
      unionbf, 4096, wt_relup, 4096, relup_b,
      out_rel, 2048, out_rel, nullptr, 2048, 8192, 2048, 4096);
}
